// flatten_GRN_20727512171033
// MI455X (gfx1250) — matrix operand order repaired and checked by reading, not compiled, not run on hardware
//
#include <hip/hip_runtime.h>
#include <stdint.h>

typedef __attribute__((ext_vector_type(16))) __bf16    v16bf;
typedef __attribute__((ext_vector_type(8)))  float     v8f;
typedef __attribute__((ext_vector_type(4)))  unsigned  u32x4;

#define BS      4
#define SEQ     1024
#define NVAR    32
#define DMODEL  512
#define DHID    256
#define NTOK    (BS*SEQ*NVAR)   /* 131072 tokens */
#define NTILE   (NTOK/16)       /* 8192 tiles of 16 tokens */
#define TPB_TILES 8
#define GRID_A  (NTILE/TPB_TILES) /* 1024 blocks */

__device__ __forceinline__ unsigned short f2bf(float f) {
    unsigned u = __float_as_uint(f);
    unsigned r = u + 0x7FFFu + ((u >> 16) & 1u); // round-to-nearest-even
    return (unsigned short)(r >> 16);
}

// ---------------------------------------------------------------------------
// Kernel 0: convert w1 [DHID x DMODEL] fp32 -> bf16 row-major (once, 256KB)
// ---------------------------------------------------------------------------
__global__ void prep_w1_bf16(const float* __restrict__ w1,
                             unsigned* __restrict__ w1b_packed) {
    int i = blockIdx.x * blockDim.x + threadIdx.x; // one packed pair per thread
    float2 f = ((const float2*)w1)[i];
    w1b_packed[i] = (unsigned)f2bf(f.x) | ((unsigned)f2bf(f.y) << 16);
}

// ---------------------------------------------------------------------------
// Kernel 1: fused  res = x@w_res ;  eta1 = ELU(x@w1^T) ; eta2 = eta1@w2 ;
//           pre = res + GLU(eta2)      (per token, 131072 outputs)
//
// Block: 512 threads (16 wave32). Wave w owns hidden columns h = 16w..16w+15,
// B fragments (w1 bf16) register-resident for the whole block.
// Two software pipelines:
//   * across tiles: next tile's x rows are global-loaded into a second
//     register set while the current tile's WMMAs run;
//   * within the GEMM: A fragments use SSA-style buffers loaded 4 steps
//     ahead of their consuming v_wmma_f32_16x16x32_bf16.
// ---------------------------------------------------------------------------
__global__ __launch_bounds__(512, 1)
void grn_main(const float* __restrict__ x,
              const float* __restrict__ w_res,
              const unsigned short* __restrict__ w1b,
              const float* __restrict__ w2,
              const float* __restrict__ gw1p, const float* __restrict__ gb1p,
              const float* __restrict__ gw2p, const float* __restrict__ gb2p,
              float* __restrict__ pre) {
    __shared__ __align__(16) unsigned short ldsA[16][DMODEL]; // 16KB bf16 tile
    __shared__ float ldsRes[16];
    __shared__ float ldsEta2[16];

    const int tid  = threadIdx.x;
    const int wave = tid >> 5;     // 0..15
    const int lane = tid & 31;
    const int l16  = lane & 15;
    const bool hi  = (lane >= 16);

    // staging role: per-lane w_res slice, columns lane*16 .. lane*16+15
    float wres[16];
    {
        const float4* p = (const float4*)w_res + lane * 4;
        #pragma unroll
        for (int j = 0; j < 4; ++j) {
            float4 v = p[j];
            wres[4*j+0] = v.x; wres[4*j+1] = v.y;
            wres[4*j+2] = v.z; wres[4*j+3] = v.w;
        }
    }
    // compute role: per-lane w2 scalar for hidden column h = wave*16 + l16
    const float w2v = w2[wave * 16 + l16];

    // Register-resident B fragments for this wave's n-tile.
    v16bf Breg[16];
    {
        const unsigned short* bp =
            w1b + (size_t)(wave * 16 + l16) * DMODEL + (hi ? 16 : 0);
        #pragma unroll
        for (int ks = 0; ks < 16; ++ks) {
            union { u32x4 q[2]; v16bf v; } u;
            const u32x4* q = (const u32x4*)(bp + ks * 32);
            u.q[0] = q[0];
            u.q[1] = q[1];
            Breg[ks] = u.v;
        }
    }

    const float gw1 = *gw1p, gb1 = *gb1p, gw2 = *gw2p, gb2 = *gb2p;

    const int tile0 = blockIdx.x * TPB_TILES;

    // ---- prologue: load tile 0's x row for this wave into registers -------
    float f[16], f2[16];
    {
        const float* xrow = x + (size_t)(tile0 * 16 + wave) * DMODEL;
        const float4* p = (const float4*)xrow + lane * 4;
        #pragma unroll
        for (int j = 0; j < 4; ++j) {
            float4 v = p[j];
            f[4*j+0] = v.x; f[4*j+1] = v.y;
            f[4*j+2] = v.z; f[4*j+3] = v.w;
        }
    }

    for (int t = 0; t < TPB_TILES; ++t) {
        const int tokenBase = (tile0 + t) * 16;
        __syncthreads(); // previous epilogue / LDS readers finished

        // ---- stage: convert f -> LDS bf16, residual dot via wave reduce ---
        {
            float rp = 0.f;
            unsigned u[8];
            #pragma unroll
            for (int j = 0; j < 8; ++j) {
                rp += f[2*j] * wres[2*j] + f[2*j+1] * wres[2*j+1];
                u[j] = (unsigned)f2bf(f[2*j]) | ((unsigned)f2bf(f[2*j+1]) << 16);
            }
            u32x4* dst = (u32x4*)&ldsA[wave][lane * 16];
            u32x4 q0 = { u[0], u[1], u[2], u[3] };
            u32x4 q1 = { u[4], u[5], u[6], u[7] };
            dst[0] = q0;
            dst[1] = q1;
            #pragma unroll
            for (int m = 16; m >= 1; m >>= 1) rp += __shfl_xor(rp, m, 32);
            if (lane == 0) ldsRes[wave] = rp;
            if (wave == 0 && lane < 16) ldsEta2[lane] = 0.f;
        }
        __syncthreads();

        // ---- kick off next tile's x loads (overlap with WMMA phase) -------
        if (t + 1 < TPB_TILES) {
            const float* xrow =
                x + (size_t)(tokenBase + 16 + wave) * DMODEL;
            const float4* p = (const float4*)xrow + lane * 4;
            #pragma unroll
            for (int j = 0; j < 4; ++j) {
                float4 v = p[j];
                f2[4*j+0] = v.x; f2[4*j+1] = v.y;
                f2[4*j+2] = v.z; f2[4*j+3] = v.w;
            }
        }

        // ---- GEMM: 16 x v_wmma_f32_16x16x32_bf16 over K=512 ---------------
        // SSA-style A buffers, loads issued 4 steps ahead of consumption.
        v8f acc = {0.f, 0.f, 0.f, 0.f, 0.f, 0.f, 0.f, 0.f};
        {
            const int m    = l16;            // A row this lane supplies
            const int kOff = hi ? 16 : 0;    // A layout: lanes 16-31 hold K+16
            union AF { u32x4 q[2]; v16bf v; };
            AF a[16];
            #pragma unroll
            for (int p = 0; p < 4; ++p) {    // prime the pipeline
                a[p].q[0] = *(const u32x4*)&ldsA[m][p * 32 + kOff];
                a[p].q[1] = *(const u32x4*)&ldsA[m][p * 32 + 8 + kOff];
            }
            #pragma unroll
            for (int ks = 0; ks < 16; ++ks) {
                if (ks + 4 < 16) {           // refill a fresh buffer first
                    a[ks+4].q[0] =
                        *(const u32x4*)&ldsA[m][(ks+4) * 32 + kOff];
                    a[ks+4].q[1] =
                        *(const u32x4*)&ldsA[m][(ks+4) * 32 + 8 + kOff];
                }
                acc = __builtin_amdgcn_wmma_f32_16x16x32_bf16(
                        false, a[ks].v, false, Breg[ks], (short)0, acc,
                        false, false);
            }
        }

        // ---- ELU + dot(w2): reduce over this wave's 16 hidden columns -----
        float ep[8];
        #pragma unroll
        for (int r = 0; r < 8; ++r) {
            float e = acc[r];
            e = (e > 0.f) ? e : (__expf(e) - 1.f);  // ELU(alpha=1)
            ep[r] = e * w2v;
        }
        #pragma unroll
        for (int r = 0; r < 8; ++r) {
            #pragma unroll
            for (int msk = 8; msk >= 1; msk >>= 1)
                ep[r] += __shfl_xor(ep[r], msk, 32); // within 16-lane half
        }
        if (l16 == 0) { // lanes 0 (rows 0..7) and 16 (rows 8..15)
            #pragma unroll
            for (int r = 0; r < 8; ++r)
                atomicAdd(&ldsEta2[r + (hi ? 8 : 0)], ep[r]); // ds_add_f32
        }
        __syncthreads();

        // ---- epilogue: GLU + residual, one lane per token -----------------
        if (wave == 0 && lane < 16) {
            float e2 = ldsEta2[lane];
            float rs = ldsRes[lane];
            float g  = (1.f / (1.f + __expf(-(e2 * gw1 + gb1))))
                       * (e2 * gw2 + gb2);
            pre[tokenBase + lane] = rs + g;
        }

        // rotate the cross-tile double buffer
        #pragma unroll
        for (int j = 0; j < 16; ++j) f[j] = f2[j];
    }
}

// ---------------------------------------------------------------------------
// Kernel 2: LayerNorm over NVAR=32 + softmax over NVAR. One wave32 per (b,s).
// ---------------------------------------------------------------------------
__global__ void ln_softmax(const float* __restrict__ pre,
                           const float* __restrict__ ln_g,
                           const float* __restrict__ ln_b,
                           float* __restrict__ out) {
    const int wave = threadIdx.x >> 5;
    const int lane = threadIdx.x & 31;
    const int grp  = blockIdx.x * 8 + wave; // 4096 groups

    float v = pre[grp * 32 + lane];
    float s = v;
    #pragma unroll
    for (int m = 16; m >= 1; m >>= 1) s += __shfl_xor(s, m, 32);
    float mu = s * (1.f / 32.f);
    float d  = v - mu;
    float vs = d * d;
    #pragma unroll
    for (int m = 16; m >= 1; m >>= 1) vs += __shfl_xor(vs, m, 32);
    float rstd = rsqrtf(vs * (1.f / 32.f) + 1e-5f);
    float y = d * rstd * ln_g[lane] + ln_b[lane];

    float mx = y;
    #pragma unroll
    for (int m = 16; m >= 1; m >>= 1) mx = fmaxf(mx, __shfl_xor(mx, m, 32));
    float e  = __expf(y - mx);
    float es = e;
    #pragma unroll
    for (int m = 16; m >= 1; m >>= 1) es += __shfl_xor(es, m, 32);
    out[grp * 32 + lane] = e / es;
}

// ---------------------------------------------------------------------------
extern "C" void kernel_launch(void* const* d_in, const int* in_sizes, int n_in,
                              void* d_out, int out_size, void* d_ws,
                              size_t ws_size, hipStream_t stream) {
    const float* x     = (const float*)d_in[0];
    const float* w_res = (const float*)d_in[1];
    const float* w1    = (const float*)d_in[2];
    const float* w2    = (const float*)d_in[3];
    const float* gw1   = (const float*)d_in[4];
    const float* gb1   = (const float*)d_in[5];
    const float* gw2   = (const float*)d_in[6];
    const float* gb2   = (const float*)d_in[7];
    const float* ln_g  = (const float*)d_in[8];
    const float* ln_b  = (const float*)d_in[9];

    unsigned short* w1b = (unsigned short*)d_ws;                 // 256 KB
    float* pre = (float*)((char*)d_ws +
                          (size_t)DHID * DMODEL * sizeof(unsigned short));

    prep_w1_bf16<<<(DHID * DMODEL / 2) / 256, 256, 0, stream>>>(
        w1, (unsigned*)w1b);
    grn_main<<<GRID_A, 512, 0, stream>>>(x, w_res, w1b, w2,
                                         gw1, gb1, gw2, gb2, pre);
    ln_softmax<<<(BS * SEQ) / 8, 256, 0, stream>>>(pre, ln_g, ln_b,
                                                   (float*)d_out);
}
